// CAdapter_12292196402086
// MI455X (gfx1250) — matrix operand order repaired and checked by reading, not compiled, not run on hardware
//
#include <hip/hip_runtime.h>

typedef __bf16 v8bf  __attribute__((ext_vector_type(8)));
typedef __bf16 v16bf __attribute__((ext_vector_type(16)));
typedef float  v8f   __attribute__((ext_vector_type(8)));
typedef unsigned int v4u __attribute__((ext_vector_type(4)));
typedef unsigned int v8u __attribute__((ext_vector_type(8)));

#define C_DIM 1000
#define H_DIM 128
#define CPAD  1024
#define ROWS  16
#define LSTRIDE 1000   // s_log row stride (TDM packs rows contiguously)

// ---- LDS layout (bytes) ----
#define OFF_LOGITS  0                       // f32 [16][1000] (TDM-packed)
#define OFF_KEYS    (OFF_LOGITS + 65536)    // f32 [16][1024]  sort keys -> dead after diffs
#define OFF_VBUF    (OFF_KEYS + 65536)      // f32 [16][1024]  diffs -> v -> suffix sums
#define OFF_PROBA   (OFF_VBUF + 65536)      // bf16 [16][1024] A-matrix (original order)
#define OFF_H1      (OFF_PROBA + 32768)     // bf16 [16][128]
#define OFF_H2      (OFF_H1 + 4096)         // bf16 [16][128]
#define OFF_IDX     (OFF_H2 + 4096)         // u16 [16][1024]  sort payload
#define OFF_CSUM    (OFF_IDX + 32768)       // f32 [16][16]    chunk sums for suffix scan
#define LDS_BYTES   (OFF_CSUM + 1024)       // 271,360 B < 320 KB per workgroup

__device__ __forceinline__ v8f wmma_bf16(v16bf a, v16bf b, v8f c) {
    // v_wmma_f32_16x16x32_bf16: D = A(16x32 bf16) x B(32x16 bf16) + C(16x16 f32)
    return __builtin_amdgcn_wmma_f32_16x16x32_bf16(
        /*neg_a=*/false, a, /*neg_b=*/false, b,
        /*c_mod=*/(short)0, c, /*reuse_a=*/false, /*reuse_b=*/false);
}

// A-fragment (16-bit, 16x32): lane L<16 holds row M=L, K = {kb..kb+7, kb+16..kb+23};
// lanes>=16 hold same rows with K shifted by +8. Two contiguous 16B chunks.
__device__ __forceinline__ v16bf load_afrag(const __bf16* mat, int rowStride,
                                            int rowM, int kbase, int hi) {
    const __bf16* p = mat + rowM * rowStride + kbase + hi * 8;
    union { v16bf v; v8bf h[2]; } u;
    u.h[0] = *(const v8bf*)(p);
    u.h[1] = *(const v8bf*)(p + 16);
    return u.v;
}

// Weight prep: convert f32 row-major W[Ktot][N] into bf16 WMMA B-fragment layout.
// B-fragment (16-bit, 32x16): lane n: col = nt*16 + (n&15); element j -> K = kt*32 + (j&7) + (j>=8 ? 16 : 0) + (n>=16 ? 8 : 0)
__global__ void prep_wmma_b(const float* __restrict__ W, __bf16* __restrict__ dst,
                            int Ktot, int N, int numKt, int numNt) {
    int gid = blockIdx.x * blockDim.x + threadIdx.x;
    int total = numKt * numNt * 512;
    if (gid >= total) return;
    int within = gid & 511;
    int frag   = gid >> 9;          // frag = nt*numKt + kt
    int kt = frag % numKt;
    int nt = frag / numKt;
    int lane = within >> 4;
    int j    = within & 15;
    int k   = kt * 32 + (j & 7) + ((j >> 3) << 4) + ((lane >= 16) ? 8 : 0);
    int col = nt * 16 + (lane & 15);
    float v = (k < Ktot && col < N) ? W[(size_t)k * N + col] : 0.0f;
    dst[gid] = (__bf16)v;
}

__global__ __launch_bounds__(256, 1)
void cadapter_kernel(const float* __restrict__ logits_g,
                     const float* __restrict__ b1,
                     const float* __restrict__ b2,
                     const float* __restrict__ b3,
                     const __bf16* __restrict__ w1f,   // 32 kt x 8 nt frags
                     const __bf16* __restrict__ w2f,   //  4 kt x 8 nt frags
                     const __bf16* __restrict__ w3f,   //  4 kt x 63 nt frags
                     float* __restrict__ out) {
    extern __shared__ char smem[];
    float*          s_log  = (float*)(smem + OFF_LOGITS);
    float*          s_key  = (float*)(smem + OFF_KEYS);
    float*          s_v    = (float*)(smem + OFF_VBUF);
    __bf16*         s_pa   = (__bf16*)(smem + OFF_PROBA);
    __bf16*         s_h1   = (__bf16*)(smem + OFF_H1);
    __bf16*         s_h2   = (__bf16*)(smem + OFF_H2);
    unsigned short* s_idx  = (unsigned short*)(smem + OFF_IDX);
    float*          s_cs   = (float*)(smem + OFF_CSUM);

    const int tid  = threadIdx.x;
    const int lane = tid & 31;
    const int wv   = tid >> 5;
    const int hi   = lane >> 4;   // 0/1 half of wave
    const int ln   = lane & 15;
    const int rowbase = blockIdx.x * ROWS;

    // ---------------- Phase 0: TDM DMA of the 16x1000 f32 logits tile into LDS ----------------
    // D# group 0: count=1 | lds_addr | 57-bit global tile addr | type=2 ("image")
    // D# group 1: data_size=4B | tensor_dim0=1000 | tensor_dim1=16 | tile=1000x16 | stride0=1000
    if (wv == 0) {
        unsigned long long ga =
            (unsigned long long)(uintptr_t)(logits_g + (size_t)rowbase * C_DIM);
        unsigned int ldsa = (unsigned int)(uintptr_t)s_log;  // generic LDS ptr: low 32 bits = LDS offset
        v4u g0 = { 1u,                                   // count=1, user mode
                   ldsa,                                  // lds_addr [63:32]
                   (unsigned int)ga,                      // global_addr [95:64]
                   (((unsigned int)(ga >> 32)) & 0x01FFFFFFu) | 0x80000000u }; // [120:96] | type=2
        v8u g1 = { 0x20000u,                              // data_size=2 (4 bytes), mask=0
                   ((unsigned int)C_DIM) << 16,           // tensor_dim0[15:0] in [63:48]
                   ((unsigned int)ROWS) << 16,            // tensor_dim1[15:0] in [95:80]
                   ((unsigned int)C_DIM) << 16,           // tile_dim0 in [127:112]
                   (unsigned int)ROWS,                    // tile_dim1 in [143:128]
                   (unsigned int)C_DIM,                   // tensor_dim0_stride[31:0]
                   0u, 0u };
        asm volatile("tensor_load_to_lds %0, %1" :: "s"(g0), "s"(g1) : "memory");
        __builtin_amdgcn_s_wait_tensorcnt(0);
    }
    __syncthreads();

    // ---------------- Phase 1: softmax from LDS (16 threads per row) ----------------
    {
        const int row = tid >> 4;
        const int t   = tid & 15;
        float m = -3.4e38f;
        for (int c = t; c < C_DIM; c += 16)
            m = fmaxf(m, s_log[row * LSTRIDE + c]);
        #pragma unroll
        for (int sh = 8; sh; sh >>= 1) m = fmaxf(m, __shfl_xor(m, sh));
        float ssum = 0.0f;
        for (int c = t; c < C_DIM; c += 16)
            ssum += __expf(s_log[row * LSTRIDE + c] - m);
        #pragma unroll
        for (int sh = 8; sh; sh >>= 1) ssum += __shfl_xor(ssum, sh);
        float inv = 1.0f / ssum;
        for (int c = t; c < CPAD; c += 16) {
            if (c < C_DIM) {
                float p = __expf(s_log[row * LSTRIDE + c] - m) * inv;
                s_key[row * CPAD + c] = p;
                s_pa[row * CPAD + c]  = (__bf16)p;
            } else {
                s_key[row * CPAD + c] = -1.0f;   // sinks to the tail (descending)
                s_pa[row * CPAD + c]  = (__bf16)0.0f;
            }
            s_idx[row * CPAD + c] = (unsigned short)c;
        }
    }

    // ---------------- Phase 2: bitonic sort, descending, 16 rows at once ----------------
    for (int k = 2; k <= CPAD; k <<= 1) {
        for (int jj = k >> 1; jj > 0; jj >>= 1) {
            __syncthreads();
            for (int e = tid; e < ROWS * (CPAD / 2); e += 256) {
                int row = e >> 9;
                int t   = e & 511;
                int i   = ((t & ~(jj - 1)) << 1) | (t & (jj - 1));
                int p   = i | jj;
                bool up = (i & k) != 0;
                int bi = row * CPAD + i, bp = row * CPAD + p;
                float ki = s_key[bi], kp = s_key[bp];
                if ((ki < kp) == !up) {
                    s_key[bi] = kp; s_key[bp] = ki;
                    unsigned short ti = s_idx[bi];
                    s_idx[bi] = s_idx[bp]; s_idx[bp] = ti;
                }
            }
        }
    }
    __syncthreads();

    // ---------------- Phase 3: diffs into v-buffer ----------------
    for (int e = tid; e < ROWS * CPAD; e += 256) {
        int row = e >> 10, j = e & (CPAD - 1);
        float d;
        if (j < C_DIM - 1)       d = s_key[row * CPAD + j] - s_key[row * CPAD + j + 1];
        else if (j == C_DIM - 1) d = 1.0f;
        else                     d = 0.0f;       // padding: kills suffix-sum contribution
        s_v[row * CPAD + j] = d;
    }
    __syncthreads();

    // ---------------- Phase 4: layer 1  (probs[16x1024] @ W1 -> h1[16x128]) ----------------
    {
        v8f acc = {0.f,0.f,0.f,0.f,0.f,0.f,0.f,0.f};
        #pragma unroll 4
        for (int kt = 0; kt < 32; ++kt) {
            v16bf a = load_afrag(s_pa, CPAD, ln, kt * 32, hi);
            v16bf b = *(const v16bf*)(w1f + (size_t)(wv * 32 + kt) * 512 + lane * 16);
            acc = wmma_bf16(a, b, acc);
        }
        int col = wv * 16 + ln;
        float bc = b1[col];
        #pragma unroll
        for (int r = 0; r < 8; ++r) {
            int m = r + hi * 8;
            float h = fmaxf(acc[r] + bc, 0.0f);
            s_h1[m * H_DIM + col] = (__bf16)h;
        }
    }
    __syncthreads();

    // ---------------- Phase 5: layer 2  (h1 @ W2 -> h2[16x128]) ----------------
    {
        v8f acc = {0.f,0.f,0.f,0.f,0.f,0.f,0.f,0.f};
        #pragma unroll
        for (int kt = 0; kt < 4; ++kt) {
            v16bf a = load_afrag(s_h1, H_DIM, ln, kt * 32, hi);
            v16bf b = *(const v16bf*)(w2f + (size_t)(wv * 4 + kt) * 512 + lane * 16);
            acc = wmma_bf16(a, b, acc);
        }
        int col = wv * 16 + ln;
        float bc = b2[col];
        #pragma unroll
        for (int r = 0; r < 8; ++r) {
            int m = r + hi * 8;
            float h = fmaxf(acc[r] + bc, 0.0f);
            s_h2[m * H_DIM + col] = (__bf16)h;
        }
    }
    __syncthreads();

    // ---------------- Phase 6: layer 3 (h2 @ W3), sigmoid, v = diffs * cal ----------------
    for (int nt = wv; nt < 63; nt += 8) {
        v8f acc = {0.f,0.f,0.f,0.f,0.f,0.f,0.f,0.f};
        #pragma unroll
        for (int kt = 0; kt < 4; ++kt) {
            v16bf a = load_afrag(s_h2, H_DIM, ln, kt * 32, hi);
            v16bf b = *(const v16bf*)(w3f + (size_t)(nt * 4 + kt) * 512 + lane * 16);
            acc = wmma_bf16(a, b, acc);
        }
        int c = nt * 16 + ln;
        if (c < C_DIM) {
            float bc = b3[c];
            #pragma unroll
            for (int r = 0; r < 8; ++r) {
                int m = r + hi * 8;
                float cal = acc[r] + bc;
                if (c != C_DIM - 1) cal = 1.0f / (1.0f + __expf(-cal));  // sigmoid except last col
                s_v[m * CPAD + c] *= cal;
            }
        }
    }
    __syncthreads();

    // ---------------- Phase 7: suffix sum of v (16 thr/row, 64-elem chunks) ----------------
    {
        const int row = tid >> 4;
        const int t   = tid & 15;
        const int base = row * CPAD + t * 64;
        float run = 0.0f;
        for (int q = 63; q >= 0; --q) { run += s_v[base + q]; s_v[base + q] = run; }
        s_cs[row * 16 + t] = run;
        __syncthreads();
        if (t == 0) {
            float acc = 0.0f;
            for (int c = 15; c >= 0; --c) {
                float tmp = s_cs[row * 16 + c];
                s_cs[row * 16 + c] = acc;   // exclusive suffix of chunk sums
                acc += tmp;
            }
        }
        __syncthreads();
        float off = s_cs[row * 16 + t];
        for (int q = 0; q < 64; ++q) s_v[base + q] += off;
    }
    __syncthreads();

    // ---------------- Phase 8: scatter via sort order, add logits ----------------
    for (int e = tid; e < ROWS * CPAD; e += 256) {
        int row = e >> 10, j = e & (CPAD - 1);
        if (j < C_DIM) {
            int c = s_idx[row * CPAD + j];
            float val = s_v[row * CPAD + j] + s_log[row * LSTRIDE + c];
            out[(size_t)(rowbase + row) * C_DIM + c] = val;
        }
    }
}

extern "C" void kernel_launch(void* const* d_in, const int* in_sizes, int n_in,
                              void* d_out, int out_size, void* d_ws, size_t ws_size,
                              hipStream_t stream) {
    const float* logits = (const float*)d_in[0];
    const float* W1 = (const float*)d_in[1];
    const float* b1 = (const float*)d_in[2];
    const float* W2 = (const float*)d_in[3];
    const float* b2 = (const float*)d_in[4];
    const float* W3 = (const float*)d_in[5];
    const float* b3 = (const float*)d_in[6];
    float* out = (float*)d_out;

    // Workspace: bf16 WMMA B-fragment copies of W1/W2/W3 (~540 KB total)
    __bf16* w1f = (__bf16*)d_ws;                 // 32*8*512  = 131072 elems
    __bf16* w2f = w1f + 32 * 8 * 512;            //  4*8*512  =  16384 elems
    __bf16* w3f = w2f + 4 * 8 * 512;             //  4*63*512 = 129024 elems

    {
        int t1 = 32 * 8 * 512;
        prep_wmma_b<<<(t1 + 255) / 256, 256, 0, stream>>>(W1, w1f, 1000, 128, 32, 8);
        int t2 = 4 * 8 * 512;
        prep_wmma_b<<<(t2 + 255) / 256, 256, 0, stream>>>(W2, w2f, 128, 128, 4, 8);
        int t3 = 4 * 63 * 512;
        prep_wmma_b<<<(t3 + 255) / 256, 256, 0, stream>>>(W3, w3f, 128, 1000, 4, 63);
    }

    cadapter_kernel<<<32768 / ROWS, 256, LDS_BYTES, stream>>>(
        logits, b1, b2, b3, w1f, w2f, w3f, out);
}
